// NAMTuringJump_55138790146724
// MI455X (gfx1250) — matrix operand order repaired and checked by reading, not compiled, not run on hardware
//
#include <hip/hip_runtime.h>

// ---------------------------------------------------------------------------
// NAM Turing-jump forward for MI455X (gfx1250, wave32, WMMA).
// Key algebraic optimization: softmax(tape @ Wk^T @ q) == softmax(tape @ (q@Wk))
// (shift-invariant), so the 550-GFLOP in-scan keys GEMM collapses to a single
// precomputed qk GEMM. All dense layers use v_wmma_f32_16x16x32_bf16.
// ---------------------------------------------------------------------------

typedef __bf16  v16bf __attribute__((ext_vector_type(16)));
typedef float   v8f   __attribute__((ext_vector_type(8)));

#define S_   512
#define B_   64
#define DIM_ 512
#define H_   256
#define G3_  768
#define NT_  4
#define HD_  128
#define L_   128
#define SB_  (S_ * B_)          // 32768
#define SBH_ (SB_ * NT_)        // 131072 heads*time

// -------------------------- WMMA helpers -----------------------------------

__device__ __forceinline__ v8f wmma_bf16(v16bf a, v16bf b, v8f c) {
  return __builtin_amdgcn_wmma_f32_16x16x32_bf16(
      /*neg_a=*/false, a, /*neg_b=*/false, b,
      /*c_mod=*/(short)0, c, /*reuse_a=*/false, /*reuse_b=*/false);
}

// A fragment: 16x32 bf16, row-major source with leading dim ldA.
// ISA 7.12.2: lanes 0-15 row M=lane, K in {0..7,16..23}; lanes 16-31 K in {8..15,24..31}.
__device__ __forceinline__ v16bf load_a_frag(const __bf16* A, int ldA,
                                             int row0, int k0, int lane) {
  int m  = row0 + (lane & 15);
  int hi = lane >> 4;
  const __bf16* p = A + (size_t)m * ldA + k0;
  v16bf a;
#pragma unroll
  for (int j = 0; j < 16; ++j) {
    int k = ((j >> 3) << 4) + (hi << 3) + (j & 7);
    a[j] = p[k];
  }
  return a;
}

// B fragment: 32x16 bf16 where B[k][n] = W[n][k] (W row-major NxK, ld=ldW).
// lanes 0-15: K in {0..7,16..23} of column n=lane; lanes 16-31: K in {8..15,24..31}.
__device__ __forceinline__ v16bf load_b_frag(const __bf16* W, int ldW,
                                             int k0, int col0, int lane) {
  int n  = col0 + (lane & 15);
  int hi = lane >> 4;
  const __bf16* p = W + (size_t)n * ldW + k0;
  v16bf b;
#pragma unroll
  for (int j = 0; j < 16; ++j) b[j] = p[((j >> 3) << 4) + (hi << 3) + (j & 7)];
  return b;
}

// -------------------------- utility kernels --------------------------------

__global__ void cast_f32_bf16_kernel(const float* __restrict__ src,
                                     __bf16* __restrict__ dst, int n) {
  int i = blockIdx.x * blockDim.x + threadIdx.x;
  if (i < n) dst[i] = (__bf16)src[i];
}

// zero-pad rows (used for action_W: 40 rows -> 64 rows)
__global__ void pad_cast_kernel(const float* __restrict__ src,
                                __bf16* __restrict__ dst,
                                int rows, int cols, int padRows) {
  int i = blockIdx.x * blockDim.x + threadIdx.x;
  if (i < padRows * cols) {
    int r = i / cols;
    dst[i] = (r < rows) ? (__bf16)src[i] : (__bf16)0.f;
  }
}

// WkT[d][c] = Wk[c][d]  (128x128)
__global__ void transpose_cast_wk_kernel(const float* __restrict__ src,
                                         __bf16* __restrict__ dst) {
  int i = blockIdx.x * blockDim.x + threadIdx.x;
  if (i < HD_ * HD_) {
    int d = i / HD_, c = i % HD_;
    dst[i] = (__bf16)src[c * HD_ + d];
  }
}

// per-head action post-processing: softmax[0:4], softmax[4:8], sigmoid[8],[9]
__global__ void act_post_kernel(float* __restrict__ acts) {
  int h = blockIdx.x * blockDim.x + threadIdx.x;
  if (h >= SBH_) return;
  float* a = acts + (size_t)h * 10;
#pragma unroll
  for (int g = 0; g < 2; ++g) {
    float x0 = a[g*4+0], x1 = a[g*4+1], x2 = a[g*4+2], x3 = a[g*4+3];
    float m = fmaxf(fmaxf(x0, x1), fmaxf(x2, x3));
    float e0 = __expf(x0 - m), e1 = __expf(x1 - m),
          e2 = __expf(x2 - m), e3 = __expf(x3 - m);
    float inv = 1.f / (e0 + e1 + e2 + e3);
    a[g*4+0] = e0*inv; a[g*4+1] = e1*inv; a[g*4+2] = e2*inv; a[g*4+3] = e3*inv;
  }
  a[8] = 1.f / (1.f + __expf(-a[8]));
  a[9] = 1.f / (1.f + __expf(-a[9]));
}

// -------------------------- generic WMMA GEMM ------------------------------
// C[M,N] = A[M,K](bf16) @ W[N,K]^T(bf16) + bias ; optional f32 and bf16 outputs.
// Block: 128 thr = 4 waves stacked in M; wave tile 16x64. M % 64 == 0.
// W buffer must have >= gridDim.y*64 (zero-padded) rows.

__global__ __launch_bounds__(128)
void gemm_bf16_kernel(const __bf16* __restrict__ A,
                      const __bf16* __restrict__ W,
                      const float* __restrict__ bias,
                      float* __restrict__ C,
                      __bf16* __restrict__ Cb,
                      int M, int N, int K) {
  int wave = threadIdx.x >> 5;
  int lane = threadIdx.x & 31;
  int row0 = blockIdx.x * 64 + wave * 16;
  int col0 = blockIdx.y * 64;

  v8f acc[4] = {v8f{}, v8f{}, v8f{}, v8f{}};
  for (int k0 = 0; k0 < K; k0 += 32) {
    v16bf a = load_a_frag(A, K, row0, k0, lane);
#pragma unroll
    for (int ct = 0; ct < 4; ++ct) {
      v16bf b = load_b_frag(W, K, k0, col0 + ct * 16, lane);
      acc[ct] = wmma_bf16(a, b, acc[ct]);
    }
    if (k0 + 32 < K)  // hint next A tile into cache (global_prefetch_b8)
      __builtin_prefetch(A + (size_t)(row0 + (lane & 15)) * K + k0 + 32, 0, 1);
  }

  int lane15 = lane & 15;
  int rbase  = (lane >> 4) << 3;
#pragma unroll
  for (int ct = 0; ct < 4; ++ct) {
    int n = col0 + ct * 16 + lane15;
    if (n < N) {
      float bv = bias ? bias[n] : 0.f;
#pragma unroll
      for (int r = 0; r < 8; ++r) {
        int m = row0 + rbase + r;
        float v = acc[ct][r] + bv;
        if (C)  C[(size_t)m * N + n] = v;
        if (Cb) Cb[(size_t)m * N + n] = (__bf16)v;
      }
    }
  }
}

// -------------------------- bidirectional GRU ------------------------------
// One block per direction. h (64x256) double-buffered bf16 in LDS; per step
// gh = h @ Whh^T via WMMA (each wave owns 32 hidden columns at all 3 gate
// offsets so the r/z/n gating is wave-local), then VALU nonlinearity.

__global__ __launch_bounds__(256)
void gru_kernel(const float* __restrict__ xg_f, const float* __restrict__ xg_b,
                const __bf16* __restrict__ whh_f, const __bf16* __restrict__ whh_b,
                const float* __restrict__ bhh_f, const float* __restrict__ bhh_b,
                __bf16* __restrict__ hidden /* (S,B,512) */) {
  __shared__ __bf16 hbuf[2][B_][H_];   // 2 x 32KB

  int dir  = blockIdx.x;
  const float*  xg  = dir ? xg_b  : xg_f;
  const __bf16* whh = dir ? whh_b : whh_f;
  const float*  bhh = dir ? bhh_b : bhh_f;

  int tid  = threadIdx.x;
  int wave = tid >> 5;
  int lane = tid & 31;
  int c0   = wave * 32;               // this wave's 32 hidden columns

  for (int i = tid; i < B_ * H_; i += 256) (&hbuf[0][0][0])[i] = (__bf16)0.f;
  __syncthreads();

  int p = 0;
  for (int i = 0; i < S_; ++i) {
    int t = dir ? (S_ - 1 - i) : i;
    const float*  xgt  = xg + (size_t)t * B_ * G3_;
    const __bf16* hsrc = &hbuf[p][0][0];

    for (int mt = 0; mt < 4; ++mt) {           // 4 row tiles of 16 batches
      v8f acc[3][2] = {{v8f{}, v8f{}}, {v8f{}, v8f{}}, {v8f{}, v8f{}}};
      for (int k0 = 0; k0 < H_; k0 += 32) {
        v16bf a = load_a_frag(hsrc, H_, mt * 16, k0, lane);
#pragma unroll
        for (int g = 0; g < 3; ++g)
#pragma unroll
          for (int ct = 0; ct < 2; ++ct) {
            v16bf b = load_b_frag(whh, H_, k0, g * H_ + c0 + ct * 16, lane);
            acc[g][ct] = wmma_bf16(a, b, acc[g][ct]);
          }
      }
#pragma unroll
      for (int ct = 0; ct < 2; ++ct) {
        int c  = c0 + ct * 16 + (lane & 15);
        float br = bhh[c], bz = bhh[H_ + c], bn = bhh[2 * H_ + c];
#pragma unroll
        for (int r = 0; r < 8; ++r) {
          int bb = mt * 16 + r + ((lane >> 4) << 3);   // batch row
          const float* xrow = xgt + (size_t)bb * G3_;
          float hr = acc[0][ct][r] + br;
          float hz = acc[1][ct][r] + bz;
          float hn = acc[2][ct][r] + bn;
          float rg = 1.f / (1.f + __expf(-(xrow[c] + hr)));
          float zg = 1.f / (1.f + __expf(-(xrow[H_ + c] + hz)));
          float ng = tanhf(xrow[2 * H_ + c] + rg * hn);
          float ho = (float)hbuf[p][bb][c];
          float hnew = (1.f - zg) * ng + zg * ho;
          hbuf[1 - p][bb][c] = (__bf16)hnew;
          hidden[((size_t)t * B_ + bb) * DIM_ + dir * H_ + c] = (__bf16)hnew;
        }
      }
    }
    __syncthreads();
    p ^= 1;
  }
}

// -------------------------- persistent tape scan ---------------------------
// One block per (b,nt) head; 128x128 f32 tape lives in LDS for all 512 steps.
// Per-step MV ops use 256 threads (two 64-length partial sums per output).

__global__ __launch_bounds__(256)
void tape_scan_kernel(const float* __restrict__ values,  // (S,B,NT,HD)
                      const float* __restrict__ acts,    // (S,B,NT,10) post
                      const float* __restrict__ qk,      // (S,B,NT,HD)
                      __bf16* __restrict__ outpre,       // (S,B,NT,HD)
                      float*  __restrict__ tape_out) {   // (L,B,NT,HD)
  __shared__ float tape[L_][HD_ + 1];          // +1 pad: conflict-free rows
  __shared__ float rpos[L_], wpos[L_], jl[L_], colv[HD_];
  __shared__ float part[4][128];
  __shared__ float red[128];

  int tid  = threadIdx.x;
  int head = blockIdx.x;
  int b    = head >> 2, nt = head & 3;

  for (int idx = tid; idx < L_ * HD_; idx += 256)
    tape[idx >> 7][idx & 127] = 0.f;
  if (tid < L_) { float v0 = (tid == 0) ? 1.f : 0.f; rpos[tid] = v0; wpos[tid] = v0; }
  __syncthreads();

  int c  = tid & 127;
  int hf = tid >> 7;
  int l0 = hf * 64;

  for (int t = 0; t < S_; ++t) {
    size_t base = (size_t)(t * B_ + b) * NT_ + nt;
    const float* v = values + base * HD_;
    const float* a = acts   + base * 10;
    const float* q = qk     + base * HD_;
    float dr0=a[0], dr1=a[1], dr2=a[2], dr3=a[3];
    float dw0=a[4], dw1=a[5], dw2=a[6], dw3=a[7];
    float rwr=a[8], rww=a[9];

    // oldval[c] = sum_l tape[l][c]*wpos[l]
    float s = 0.f;
    for (int l = l0; l < l0 + 64; ++l) s += tape[l][c] * wpos[l];
    part[hf][c] = s;
    __syncthreads();
    if (tid < HD_) colv[tid] = (v[tid] - (part[0][tid] + part[1][tid])) * rww;
    __syncthreads();

    // tape += wpos (x) colv
    for (int idx = tid; idx < L_ * HD_; idx += 256) {
      int l = idx >> 7, cc = idx & 127;
      tape[l][cc] += wpos[l] * colv[cc];
    }
    __syncthreads();

    // read_out partials (over l) and jlogit partials (over c)
    float sro = 0.f;
    for (int l = l0; l < l0 + 64; ++l) sro += tape[l][c] * rpos[l];
    float sjl = 0.f;
    for (int cc = l0; cc < l0 + 64; ++cc) sjl += tape[c][cc] * q[cc];
    part[hf][c]     = sro;
    part[2 + hf][c] = sjl;
    __syncthreads();
    if (tid < HD_) {
      float ro = part[0][tid] + part[1][tid];
      outpre[base * HD_ + tid] = (__bf16)(ro * rwr);
      jl[tid] = part[2][tid] + part[3][tid];
    }
    __syncthreads();

    // softmax over tape positions
    if (tid < L_) red[tid] = jl[tid];
    __syncthreads();
    for (int off = 64; off > 0; off >>= 1) {
      if (tid < off) red[tid] = fmaxf(red[tid], red[tid + off]);
      __syncthreads();
    }
    float mx = red[0];
    __syncthreads();
    if (tid < L_) { float e = __expf(jl[tid] - mx); jl[tid] = e; red[tid] = e; }
    __syncthreads();
    for (int off = 64; off > 0; off >>= 1) {
      if (tid < off) red[tid] += red[tid + off];
      __syncthreads();
    }
    float inv = 1.f / red[0];

    // position updates: nxt[l]=rpos[l-1], prv[l]=rpos[l+1] (jnp.roll semantics)
    float newr = 0.f, neww = 0.f;
    if (tid < L_) {
      float jp  = jl[tid] * inv;
      float prv = rpos[(tid + 1) & 127];
      float nxt = rpos[(tid + 127) & 127];
      float ro  = rpos[tid], wo = wpos[tid];
      neww = prv * dw0 + wo * dw1 + nxt * dw2 + jp * dw3;
      newr = prv * dr0 + ro * dr1 + nxt * dr2 + jp * dr3;
    }
    __syncthreads();
    if (tid < L_) { rpos[tid] = newr; wpos[tid] = neww; }
    __syncthreads();
  }

  // final tape -> d_out[(l,b,nt,c)]
  for (int idx = tid; idx < L_ * HD_; idx += 256) {
    int l = idx >> 7, cc = idx & 127;
    tape_out[(((size_t)l * B_ + b) * NT_ + nt) * HD_ + cc] = tape[l][cc];
  }
}

// -------------------------- host orchestration -----------------------------

extern "C" void kernel_launch(void* const* d_in, const int* in_sizes, int n_in,
                              void* d_out, int out_size, void* d_ws, size_t ws_size,
                              hipStream_t stream) {
  const float* inputs   = (const float*)d_in[0];
  const float* Wih_f    = (const float*)d_in[1];
  const float* Whh_f    = (const float*)d_in[2];
  const float* bih_f    = (const float*)d_in[3];
  const float* bhh_f    = (const float*)d_in[4];
  const float* Wih_b    = (const float*)d_in[5];
  const float* Whh_b    = (const float*)d_in[6];
  const float* bih_b    = (const float*)d_in[7];
  const float* bhh_b    = (const float*)d_in[8];
  const float* action_W = (const float*)d_in[9];
  const float* action_b = (const float*)d_in[10];
  const float* value_W  = (const float*)d_in[11];
  const float* value_b  = (const float*)d_in[12];
  const float* out_W    = (const float*)d_in[13];
  const float* out_b    = (const float*)d_in[14];
  const float* Wk_W     = (const float*)d_in[15];
  /* Wk_b (d_in[16]) is a softmax shift -> dropped */
  const float* Wq_W     = (const float*)d_in[17];
  const float* Wq_b     = (const float*)d_in[18];

  char* p = (char*)d_ws;
  auto alloc = [&](size_t bytes) -> char* {
    char* r = p; p += (bytes + 255) & ~(size_t)255; return r;
  };

  __bf16* x_bf      = (__bf16*)alloc((size_t)SB_ * DIM_ * 2);
  __bf16* wihf_bf   = (__bf16*)alloc((size_t)G3_ * DIM_ * 2);
  __bf16* wihb_bf   = (__bf16*)alloc((size_t)G3_ * DIM_ * 2);
  __bf16* whhf_bf   = (__bf16*)alloc((size_t)G3_ * H_ * 2);
  __bf16* whhb_bf   = (__bf16*)alloc((size_t)G3_ * H_ * 2);
  __bf16* valw_bf   = (__bf16*)alloc((size_t)DIM_ * DIM_ * 2);
  __bf16* actw_bf   = (__bf16*)alloc((size_t)64 * DIM_ * 2);   // padded 40->64
  __bf16* outw_bf   = (__bf16*)alloc((size_t)DIM_ * DIM_ * 2);
  __bf16* wq_bf     = (__bf16*)alloc((size_t)DIM_ * DIM_ * 2);
  __bf16* wkT_bf    = (__bf16*)alloc((size_t)HD_ * HD_ * 2);
  float*  xg_f      = (float*)alloc((size_t)SB_ * G3_ * 4);
  float*  xg_b      = (float*)alloc((size_t)SB_ * G3_ * 4);
  float*  values_f  = (float*)alloc((size_t)SB_ * DIM_ * 4);
  __bf16* hidden_bf = (__bf16*)alloc((size_t)SB_ * DIM_ * 2);
  float*  actions_f = (float*)alloc((size_t)SB_ * 40 * 4);
  __bf16* quer_bf   = (__bf16*)alloc((size_t)SB_ * DIM_ * 2);
  float*  qk_f      = (float*)alloc((size_t)SB_ * DIM_ * 4);
  __bf16* outpre_bf = (__bf16*)alloc((size_t)SB_ * DIM_ * 2);

  float* out_main = (float*)d_out;                    // (S,B,DIM)
  float* out_tape = out_main + (size_t)SB_ * DIM_;    // (L,B,DIM)

  auto cast = [&](const float* src, __bf16* dst, int n) {
    cast_f32_bf16_kernel<<<dim3((n + 255) / 256), 256, 0, stream>>>(src, dst, n);
  };
  auto gemm = [&](const __bf16* A, const __bf16* W, const float* bias,
                  float* C, __bf16* Cb, int M, int N, int K) {
    dim3 g(M / 64, (N + 63) / 64);
    gemm_bf16_kernel<<<g, 128, 0, stream>>>(A, W, bias, C, Cb, M, N, K);
  };

  // 1) cast inputs + weights to bf16
  cast(inputs, x_bf, SB_ * DIM_);
  cast(Wih_f, wihf_bf, G3_ * DIM_);
  cast(Wih_b, wihb_bf, G3_ * DIM_);
  cast(Whh_f, whhf_bf, G3_ * H_);
  cast(Whh_b, whhb_bf, G3_ * H_);
  cast(value_W, valw_bf, DIM_ * DIM_);
  cast(out_W, outw_bf, DIM_ * DIM_);
  cast(Wq_W, wq_bf, DIM_ * DIM_);
  pad_cast_kernel<<<dim3((64 * DIM_ + 255) / 256), 256, 0, stream>>>(
      action_W, actw_bf, 40, DIM_, 64);
  transpose_cast_wk_kernel<<<dim3((HD_ * HD_ + 255) / 256), 256, 0, stream>>>(
      Wk_W, wkT_bf);

  // 2) input-side GEMMs
  gemm(x_bf, wihf_bf, bih_f, xg_f, nullptr, SB_, G3_, DIM_);
  gemm(x_bf, wihb_bf, bih_b, xg_b, nullptr, SB_, G3_, DIM_);
  gemm(x_bf, valw_bf, value_b, values_f, nullptr, SB_, DIM_, DIM_);

  // 3) bidirectional GRU (WMMA recurrent matmul, LDS-resident state)
  gru_kernel<<<2, 256, 0, stream>>>(xg_f, xg_b, whhf_bf, whhb_bf,
                                    bhh_f, bhh_b, hidden_bf);

  // 4) hidden-side GEMMs + gate post-processing
  gemm(hidden_bf, actw_bf, action_b, actions_f, nullptr, SB_, 40, DIM_);
  act_post_kernel<<<dim3(SBH_ / 256), 256, 0, stream>>>(actions_f);
  gemm(hidden_bf, wq_bf, Wq_b, nullptr, quer_bf, SB_, DIM_, DIM_);

  // 5) qk = q @ Wk  (queries viewed as (S*B*NT, 128)); Wk_b dropped (softmax shift)
  gemm(quer_bf, wkT_bf, nullptr, qk_f, nullptr, SBH_, HD_, HD_);

  // 6) persistent per-head tape scan (tape lives in LDS for all 512 steps)
  tape_scan_kernel<<<dim3(B_ * NT_), 256, 0, stream>>>(
      values_f, actions_f, qk_f, outpre_bf, out_tape);

  // 7) output projection
  gemm(outpre_bf, outw_bf, out_b, out_main, nullptr, SB_, DIM_, DIM_);
}
